// TwinsGCN_23158463660766
// MI455X (gfx1250) — matrix operand order repaired and checked by reading, not compiled, not run on hardware
//
#include <hip/hip_runtime.h>
#include <hip/hip_bf16.h>

typedef unsigned short ushort_t;
typedef unsigned int uint_t;

#define N_C_   50000
#define E_C_   1600000
#define N_F_   50000
#define E_F_   250000
#define F_IN_  256
#define B_     8

typedef __attribute__((ext_vector_type(16))) __bf16 v16bf;
typedef __attribute__((ext_vector_type(8)))  float  v8f;

static __device__ __forceinline__ ushort_t f2bf(float x) {
  uint_t u = __float_as_uint(x);
  uint_t r = (u + 0x7FFFu + ((u >> 16) & 1u)) >> 16;
  return (ushort_t)r;
}

// ---------------------------------------------------------------------------
// WMMA GEMM: C[M x N] (f32) = A[M x K] (bf16, row-major) @ Bt[N x K] (bf16,
// weights pre-transposed). K multiple of 32, N multiple of 64, M multiple of 16.
// One wave -> 16(M) x 64(N) strip: 4 accumulators, A fragment reused 4x.
// Fragment gathers follow the CDNA5 ISA 16-bit A / B / f32 C VGPR tables.
// ---------------------------------------------------------------------------
template<int K>
__global__ void wmma_gemm_kernel(const ushort_t* __restrict__ A,
                                 const ushort_t* __restrict__ Bt,
                                 float* __restrict__ C,
                                 int M, int N) {
  const int lane = threadIdx.x & 31;
  const int wave = blockIdx.x * (blockDim.x >> 5) + (threadIdx.x >> 5);
  const int nquads = N >> 6;           // 64-wide N strips
  const int mtiles = M >> 4;
  if (wave >= mtiles * nquads) return;
  const int mt = wave / nquads;
  const int nq = wave % nquads;
  const int m  = lane & 15;
  const int h  = lane >> 4;

  v8f acc[4];
  #pragma unroll
  for (int t = 0; t < 4; ++t)
    #pragma unroll
    for (int r = 0; r < 8; ++r) acc[t][r] = 0.0f;

  const ushort_t* arow = A + (size_t)(mt * 16 + m) * K;

  for (int k0 = 0; k0 < K; k0 += 32) {
    // A 16x32 bf16 fragment: lane (m,h) holds K = [8h..8h+7] and [16+8h..16+8h+7]
    union { v16bf v; uint4 q[2]; } af;
    af.q[0] = *(const uint4*)(arow + k0 + 8 * h);
    af.q[1] = *(const uint4*)(arow + k0 + 16 + 8 * h);
    if (k0 + 32 < K) __builtin_prefetch((const void*)(arow + k0 + 32), 0, 1);

    #pragma unroll
    for (int t = 0; t < 4; ++t) {
      // B 32x16 bf16 fragment: lane (n=m,h) holds K = [8h..8h+7] and [16+8h..16+8h+7] of column n
      const int ncol = nq * 64 + t * 16 + m;
      const ushort_t* brow = Bt + (size_t)ncol * K + k0 + 8 * h;
      union { v16bf v; uint4 q[2]; } bf;
      bf.q[0] = *(const uint4*)(brow);
      bf.q[1] = *(const uint4*)(brow + 16);
      acc[t] = __builtin_amdgcn_wmma_f32_16x16x32_bf16(
          false, af.v, false, bf.v, (short)0, acc[t], false, false);
    }
  }

  // D layout: VGPR r -> row (r + 8h), col (lane & 15)
  #pragma unroll
  for (int t = 0; t < 4; ++t) {
    const int ncol = nq * 64 + t * 16 + m;
    #pragma unroll
    for (int r = 0; r < 8; ++r)
      C[(size_t)(mt * 16 + r + 8 * h) * N + ncol] = acc[t][r];
  }
}

// ---------------------------------------------------------------------------
// Elementwise / graph kernels
// ---------------------------------------------------------------------------
__global__ void cvt_bf16_kernel(const float* __restrict__ src,
                                ushort_t* __restrict__ dst, size_t n) {
  size_t gid = (size_t)blockIdx.x * blockDim.x + threadIdx.x;
  if (gid < n) dst[gid] = f2bf(src[gid]);
}

// W [K x N] f32 -> Wt [N x K] bf16
__global__ void wt_transpose_kernel(const float* __restrict__ W,
                                    ushort_t* __restrict__ Wt, int K, int N) {
  size_t gid = (size_t)blockIdx.x * blockDim.x + threadIdx.x;
  if (gid >= (size_t)K * N) return;
  int k = (int)(gid / N), n = (int)(gid % N);
  Wt[(size_t)n * K + k] = f2bf(W[gid]);
}

__global__ void att_softmax_kernel(const float* __restrict__ x,
                                   const float* __restrict__ W,   // 3x8
                                   const float* __restrict__ b,   // 8
                                   float* __restrict__ a, int N) {
  int n = blockIdx.x * blockDim.x + threadIdx.x;
  if (n >= N) return;
  float x0 = x[n * 3], x1 = x[n * 3 + 1], x2 = x[n * 3 + 2];
  float l[8], mx = -1e30f;
  #pragma unroll
  for (int hh = 0; hh < 8; ++hh) {
    l[hh] = x0 * W[hh] + x1 * W[8 + hh] + x2 * W[16 + hh] + b[hh];
    mx = fmaxf(mx, l[hh]);
  }
  float s = 0.f;
  #pragma unroll
  for (int hh = 0; hh < 8; ++hh) { l[hh] = __expf(l[hh] - mx); s += l[hh]; }
  float inv = 1.0f / s;
  #pragma unroll
  for (int hh = 0; hh < 8; ++hh) a[n * 8 + hh] = l[hh] * inv;
}

// tiny per-node GEMM (coord branch): h[N x NO] = x[N x K] @ W[K x NO]
template<int K, int NO>
__global__ void node_mm_kernel(const float* __restrict__ x,
                               const float* __restrict__ W,
                               float* __restrict__ h, int N) {
  size_t gid = (size_t)blockIdx.x * blockDim.x + threadIdx.x;
  if (gid >= (size_t)N * NO) return;
  int n = (int)(gid / NO), c = (int)(gid % NO);
  float acc = 0.f;
  #pragma unroll
  for (int k = 0; k < K; ++k) acc += x[(size_t)n * K + k] * W[k * NO + c];
  h[gid] = acc;
}

__global__ void deg_kernel(const int* __restrict__ dst,
                           const float* __restrict__ ew,
                           float* __restrict__ deg, int E) {
  int e = blockIdx.x * blockDim.x + threadIdx.x;
  if (e < E) atomicAdd(&deg[dst[e]], ew[e]);
}

__global__ void dis_finalize_kernel(float* __restrict__ deg, int N) {
  int n = blockIdx.x * blockDim.x + threadIdx.x;
  if (n < N) deg[n] = rsqrtf(deg[n] + 1.0f);   // self-loop weight 1; deg >= 1
}

template<int C>
__global__ void scatter_gcn_kernel(const float* __restrict__ h,
                                   const int* __restrict__ src,
                                   const int* __restrict__ dst,
                                   const float* __restrict__ ew,
                                   const float* __restrict__ dis,
                                   float* __restrict__ agg, int E) {
  size_t gid = (size_t)blockIdx.x * blockDim.x + threadIdx.x;
  if (gid >= (size_t)E * C) return;
  int e = (int)(gid / C), c = (int)(gid % C);
  int s = src[e], d = dst[e];
  float norm = dis[s] * ew[e] * dis[d];
  atomicAdd(&agg[(size_t)d * C + c], h[(size_t)s * C + c] * norm);
}

// Fused: combined = relu(agg + h*dis^2 + bias); out = LN(combined)*g + b.
// One wave per row; optional bf16 mirror for the next WMMA GEMM.
template<int C>
__global__ void combine_ln_kernel(const float* __restrict__ agg,
                                  const float* __restrict__ h,
                                  const float* __restrict__ dis,
                                  const float* __restrict__ bias,
                                  const float* __restrict__ gamma,
                                  const float* __restrict__ beta,
                                  float* __restrict__ out,
                                  ushort_t* __restrict__ out_bf, int N) {
  const int lane = threadIdx.x & 31;
  const int row  = blockIdx.x * (blockDim.x >> 5) + (threadIdx.x >> 5);
  if (row >= N) return;
  const float sw = dis[row] * dis[row];
  constexpr int PER = (C + 31) / 32;
  float vals[PER];
  float sum = 0.f;
  #pragma unroll
  for (int i = 0; i < PER; ++i) {
    int c = lane + i * 32;
    float v = 0.f;
    if (c < C) {
      v = agg[(size_t)row * C + c] + h[(size_t)row * C + c] * sw + bias[c];
      v = fmaxf(v, 0.f);
    }
    vals[i] = v; sum += v;
  }
  #pragma unroll
  for (int off = 16; off > 0; off >>= 1) sum += __shfl_xor(sum, off, 32);
  float mean = sum * (1.0f / C);
  float var = 0.f;
  #pragma unroll
  for (int i = 0; i < PER; ++i) {
    int c = lane + i * 32;
    float d = (c < C) ? (vals[i] - mean) : 0.f;
    var += d * d;
  }
  #pragma unroll
  for (int off = 16; off > 0; off >>= 1) var += __shfl_xor(var, off, 32);
  float inv = rsqrtf(var * (1.0f / C) + 1e-5f);
  #pragma unroll
  for (int i = 0; i < PER; ++i) {
    int c = lane + i * 32;
    if (c < C) {
      float y = (vals[i] - mean) * inv * gamma[c] + beta[c];
      out[(size_t)row * C + c] = y;
      if (out_bf) out_bf[(size_t)row * C + c] = f2bf(y);
    }
  }
}

template<int C>
__global__ void pool_sum_kernel(const float* __restrict__ x,
                                const int* __restrict__ batch,
                                float* __restrict__ s, float* __restrict__ cnt,
                                int N) {
  size_t gid = (size_t)blockIdx.x * blockDim.x + threadIdx.x;
  if (gid >= (size_t)N * C) return;
  int n = (int)(gid / C), c = (int)(gid % C);
  int b = batch[n];
  atomicAdd(&s[b * C + c], x[gid]);
  if (c == 0) atomicAdd(&cnt[b], 1.0f);
}

__global__ void pool_div_kernel(float* __restrict__ s,
                                const float* __restrict__ cnt, int total, int C) {
  int gid = blockIdx.x * blockDim.x + threadIdx.x;
  if (gid < total) s[gid] /= fmaxf(cnt[gid / C], 1.0f);
}

// one-block tiny GEMM: C[M x N] = (relu?)(A[M x K] @ W[K x N] + bias)
__global__ void small_gemm_kernel(const float* __restrict__ A,
                                  const float* __restrict__ W,
                                  const float* __restrict__ bias,
                                  float* __restrict__ C,
                                  int M, int N, int K, int do_relu) {
  for (int i = threadIdx.x; i < M * N; i += blockDim.x) {
    int m = i / N, n = i % N;
    float acc = bias ? bias[n] : 0.f;
    for (int k = 0; k < K; ++k) acc += A[m * K + k] * W[k * N + n];
    C[i] = do_relu ? fmaxf(acc, 0.f) : acc;
  }
}

// out[8x4] = concat(cg, fg)[8x256] @ fc_W[256x4] + fc_b
__global__ void final_out_kernel(const float* __restrict__ cg,
                                 const float* __restrict__ fg,
                                 const float* __restrict__ W,
                                 const float* __restrict__ b,
                                 float* __restrict__ out) {
  int i = threadIdx.x;
  if (i >= 32) return;
  int m = i / 4, n = i % 4;
  float acc = b[n];
  for (int k = 0; k < 128; ++k) acc += cg[m * 128 + k] * W[k * 4 + n];
  for (int k = 0; k < 128; ++k) acc += fg[m * 128 + k] * W[(128 + k) * 4 + n];
  out[i] = acc;
}

// ---------------------------------------------------------------------------
static inline unsigned cdiv(size_t a, size_t b) { return (unsigned)((a + b - 1) / b); }

extern "C" void kernel_launch(void* const* d_in, const int* in_sizes, int n_in,
                              void* d_out, int out_size, void* d_ws, size_t ws_size,
                              hipStream_t stream) {
  const float* coord_x     = (const float*)d_in[0];
  const float* feature_x   = (const float*)d_in[1];
  const float* coord_ew    = (const float*)d_in[2];
  const float* feat_ew     = (const float*)d_in[3];
  const int*   coord_src   = (const int*)d_in[4];
  const int*   coord_dst   = (const int*)d_in[5];
  const int*   feat_src    = (const int*)d_in[6];
  const int*   feat_dst    = (const int*)d_in[7];
  const int*   coord_batch = (const int*)d_in[8];
  const int*   feat_batch  = (const int*)d_in[9];
  const float* att_W  = (const float*)d_in[10]; const float* att_b  = (const float*)d_in[11];
  const float* c1_W   = (const float*)d_in[12]; const float* c1_b   = (const float*)d_in[13];
  const float* cln1_g = (const float*)d_in[14]; const float* cln1_b = (const float*)d_in[15];
  const float* c2_W   = (const float*)d_in[16]; const float* c2_b   = (const float*)d_in[17];
  const float* cln2_g = (const float*)d_in[18]; const float* cln2_b = (const float*)d_in[19];
  const float* cfc_W  = (const float*)d_in[20]; const float* cfc_b  = (const float*)d_in[21];
  const float* f1_W   = (const float*)d_in[22]; const float* f1_b   = (const float*)d_in[23];
  const float* fln1_g = (const float*)d_in[24]; const float* fln1_b = (const float*)d_in[25];
  const float* f2_W   = (const float*)d_in[26]; const float* f2_b   = (const float*)d_in[27];
  const float* fln2_g = (const float*)d_in[28]; const float* fln2_b = (const float*)d_in[29];
  const float* ffc_W  = (const float*)d_in[30]; const float* ffc_b  = (const float*)d_in[31];
  const float* cp1_W  = (const float*)d_in[32]; const float* cp1_b  = (const float*)d_in[33];
  const float* cp2_W  = (const float*)d_in[34]; const float* cp2_b  = (const float*)d_in[35];
  const float* fp1_W  = (const float*)d_in[36]; const float* fp1_b  = (const float*)d_in[37];
  const float* fp2_W  = (const float*)d_in[38]; const float* fp2_b  = (const float*)d_in[39];
  const float* fc_W   = (const float*)d_in[40]; const float* fc_b   = (const float*)d_in[41];

  // ---- workspace carve ----
  char* w = (char*)d_ws;
  auto carve = [&](size_t bytes) { char* p = w; w += (bytes + 255) & ~(size_t)255; return p; };
  float*    a_att  = (float*)carve((size_t)N_C_ * 8 * 4);
  float*    dis_c  = (float*)carve((size_t)N_C_ * 4);
  float*    h_c    = (float*)carve((size_t)N_C_ * 32 * 4);
  float*    agg_c  = (float*)carve((size_t)N_C_ * 32 * 4);
  float*    x1_c   = (float*)carve((size_t)N_C_ * 16 * 4);
  float*    dis_f  = (float*)carve((size_t)N_F_ * 4);
  ushort_t* Xb     = (ushort_t*)carve((size_t)N_F_ * 512 * 2);
  ushort_t* Wt     = (ushort_t*)carve((size_t)512 * 512 * 2);
  float*    bufH   = (float*)carve((size_t)N_F_ * 512 * 4);
  float*    bufAgg = (float*)carve((size_t)N_F_ * 512 * 4);
  float*    pool_c = (float*)carve((size_t)B_ * 32 * 4);
  float*    cnt_c  = (float*)carve((size_t)B_ * 4);
  float*    pool_f = (float*)carve((size_t)B_ * 256 * 4);
  float*    cnt_f  = (float*)carve((size_t)B_ * 4);
  float*    cg     = (float*)carve((size_t)B_ * 128 * 4);
  float*    fg     = (float*)carve((size_t)B_ * 128 * 4);
  float*    thead  = (float*)carve((size_t)B_ * 128 * 4);

  // ---- output carve (return order: out, cn, fn, cproj, fproj) ----
  float* out_ptr   = (float*)d_out;
  float* cn_ptr    = out_ptr + 32;
  float* fn_ptr    = cn_ptr + (size_t)N_C_ * 32;
  float* cproj_ptr = fn_ptr + (size_t)N_F_ * 256;
  float* fproj_ptr = cproj_ptr + 8 * 64;

  const int T = 256;

  // ================= coord branch =================
  hipMemsetAsync(dis_c, 0, (size_t)N_C_ * 4, stream);
  att_softmax_kernel<<<cdiv(N_C_, T), T, 0, stream>>>(coord_x, att_W, att_b, a_att, N_C_);
  node_mm_kernel<8, 16><<<cdiv((size_t)N_C_ * 16, T), T, 0, stream>>>(a_att, c1_W, h_c, N_C_);
  deg_kernel<<<cdiv(E_C_, T), T, 0, stream>>>(coord_dst, coord_ew, dis_c, E_C_);
  dis_finalize_kernel<<<cdiv(N_C_, T), T, 0, stream>>>(dis_c, N_C_);

  hipMemsetAsync(agg_c, 0, (size_t)N_C_ * 16 * 4, stream);
  scatter_gcn_kernel<16><<<cdiv((size_t)E_C_ * 16, T), T, 0, stream>>>(
      h_c, coord_src, coord_dst, coord_ew, dis_c, agg_c, E_C_);
  combine_ln_kernel<16><<<cdiv(N_C_, 8), T, 0, stream>>>(
      agg_c, h_c, dis_c, c1_b, cln1_g, cln1_b, x1_c, nullptr, N_C_);

  node_mm_kernel<16, 32><<<cdiv((size_t)N_C_ * 32, T), T, 0, stream>>>(x1_c, c2_W, h_c, N_C_);
  hipMemsetAsync(agg_c, 0, (size_t)N_C_ * 32 * 4, stream);
  scatter_gcn_kernel<32><<<cdiv((size_t)E_C_ * 32, T), T, 0, stream>>>(
      h_c, coord_src, coord_dst, coord_ew, dis_c, agg_c, E_C_);
  combine_ln_kernel<32><<<cdiv(N_C_, 8), T, 0, stream>>>(
      agg_c, h_c, dis_c, c2_b, cln2_g, cln2_b, cn_ptr, nullptr, N_C_);

  hipMemsetAsync(pool_c, 0, (size_t)B_ * 32 * 4, stream);
  hipMemsetAsync(cnt_c, 0, (size_t)B_ * 4, stream);
  pool_sum_kernel<32><<<cdiv((size_t)N_C_ * 32, T), T, 0, stream>>>(cn_ptr, coord_batch, pool_c, cnt_c, N_C_);
  pool_div_kernel<<<1, 256, 0, stream>>>(pool_c, cnt_c, B_ * 32, 32);
  small_gemm_kernel<<<1, 256, 0, stream>>>(pool_c, cfc_W, cfc_b, cg, B_, 128, 32, 0);

  // ================= feature branch =================
  cvt_bf16_kernel<<<cdiv((size_t)N_F_ * 256, T), T, 0, stream>>>(feature_x, Xb, (size_t)N_F_ * 256);
  wt_transpose_kernel<<<cdiv((size_t)256 * 512, T), T, 0, stream>>>(f1_W, Wt, 256, 512);
  // h1 = X @ W1 : M=50000, K=256, N=512 -> 3125 mtiles * 8 nquads = 25000 waves
  wmma_gemm_kernel<256><<<cdiv((size_t)(N_F_ / 16) * (512 / 64) * 32, T), T, 0, stream>>>(
      Xb, Wt, bufH, N_F_, 512);

  hipMemsetAsync(dis_f, 0, (size_t)N_F_ * 4, stream);
  deg_kernel<<<cdiv(E_F_, T), T, 0, stream>>>(feat_dst, feat_ew, dis_f, E_F_);
  dis_finalize_kernel<<<cdiv(N_F_, T), T, 0, stream>>>(dis_f, N_F_);

  hipMemsetAsync(bufAgg, 0, (size_t)N_F_ * 512 * 4, stream);
  scatter_gcn_kernel<512><<<cdiv((size_t)E_F_ * 512, T), T, 0, stream>>>(
      bufH, feat_src, feat_dst, feat_ew, dis_f, bufAgg, E_F_);
  // fused bias+self-loop+relu+LN, f32 in-place over agg, bf16 mirror for GEMM2
  combine_ln_kernel<512><<<cdiv(N_F_, 8), T, 0, stream>>>(
      bufAgg, bufH, dis_f, f1_b, fln1_g, fln1_b, bufAgg, Xb, N_F_);

  wt_transpose_kernel<<<cdiv((size_t)512 * 256, T), T, 0, stream>>>(f2_W, Wt, 512, 256);
  // h2 = x1 @ W2 : M=50000, K=512, N=256 -> 12500 waves
  wmma_gemm_kernel<512><<<cdiv((size_t)(N_F_ / 16) * (256 / 64) * 32, T), T, 0, stream>>>(
      Xb, Wt, bufH, N_F_, 256);

  hipMemsetAsync(bufAgg, 0, (size_t)N_F_ * 256 * 4, stream);
  scatter_gcn_kernel<256><<<cdiv((size_t)E_F_ * 256, T), T, 0, stream>>>(
      bufH, feat_src, feat_dst, feat_ew, dis_f, bufAgg, E_F_);
  combine_ln_kernel<256><<<cdiv(N_F_, 8), T, 0, stream>>>(
      bufAgg, bufH, dis_f, f2_b, fln2_g, fln2_b, fn_ptr, nullptr, N_F_);

  hipMemsetAsync(pool_f, 0, (size_t)B_ * 256 * 4, stream);
  hipMemsetAsync(cnt_f, 0, (size_t)B_ * 4, stream);
  pool_sum_kernel<256><<<cdiv((size_t)N_F_ * 256, T), T, 0, stream>>>(fn_ptr, feat_batch, pool_f, cnt_f, N_F_);
  pool_div_kernel<<<1, 256, 0, stream>>>(pool_f, cnt_f, B_ * 256, 256);
  small_gemm_kernel<<<1, 256, 0, stream>>>(pool_f, ffc_W, ffc_b, fg, B_, 128, 256, 0);

  // ================= heads =================
  small_gemm_kernel<<<1, 256, 0, stream>>>(cg, cp1_W, cp1_b, thead, B_, 128, 128, 1);
  small_gemm_kernel<<<1, 256, 0, stream>>>(thead, cp2_W, cp2_b, cproj_ptr, B_, 64, 128, 0);
  small_gemm_kernel<<<1, 256, 0, stream>>>(fg, fp1_W, fp1_b, thead, B_, 128, 128, 1);
  small_gemm_kernel<<<1, 256, 0, stream>>>(thead, fp2_W, fp2_b, fproj_ptr, B_, 64, 128, 0);
  final_out_kernel<<<1, 64, 0, stream>>>(cg, fg, fc_W, fc_b, out_ptr);
}
